// StrictDualCausalTransformerBlock_4243427688655
// MI455X (gfx1250) — matrix operand order repaired and checked by reading, not compiled, not run on hardware
//
#include <hip/hip_runtime.h>
#include <hip/hip_bf16.h>

// ---------------------------------------------------------------------------
// Types for CDNA5 WMMA (wave32, v_wmma_f32_16x16x32_bf16)
// ---------------------------------------------------------------------------
typedef __bf16 bf16_t;
typedef __attribute__((ext_vector_type(16))) __bf16 v16bf;
typedef __attribute__((ext_vector_type(8)))  __bf16 v8bf;
typedef __attribute__((ext_vector_type(8)))  float  v8f;

union V16U { v16bf v; v8bf h[2]; };

__device__ inline v8f vzero8() {
  v8f v;
#pragma unroll
  for (int i = 0; i < 8; ++i) v[i] = 0.0f;
  return v;
}

__device__ inline v8f wmma_bf16(v16bf a, v16bf b, v8f c) {
  // D = A(16x32 bf16) * B(32x16 bf16) + C(16x16 f32)
  return __builtin_amdgcn_wmma_f32_16x16x32_bf16(
      /*neg_a=*/false, a, /*neg_b=*/false, b,
      /*c_mod=*/(short)0, c, /*reuse_a=*/false, /*reuse_b=*/false);
}

#define D_   1024
#define L_   1024
#define B_   4
#define H_   16
#define HD_  64
#define FF_  4096
#define KVR  1025   /* rows per batch in kv layout: cond token + L */

// ---------------------------------------------------------------------------
// fp32 -> bf16 weight conversion
// ---------------------------------------------------------------------------
__global__ void k_cvt(const float* __restrict__ src, bf16_t* __restrict__ dst, int n) {
  int i = blockIdx.x * blockDim.x + threadIdx.x;
  if (i < n) dst[i] = (bf16_t)src[i];
}

// ---------------------------------------------------------------------------
// AdaLN scale/shift GEMV: out[p][b][o] = cond[b] . W_p[o] + bias_p[o]
// ---------------------------------------------------------------------------
struct AdalnPtrs { const float* w[6]; const float* b[6]; };

__global__ void k_adaln_mod(AdalnPtrs P, const float* __restrict__ cond,
                            float* __restrict__ out) {
  int id = blockIdx.x * blockDim.x + threadIdx.x;   // 6*4*1024
  int o = id & (D_ - 1);
  int b = (id >> 10) & 3;
  int p = id >> 12;
  const float* w = P.w[p] + (size_t)o * D_;
  const float* c = cond + (size_t)b * D_;
  float acc = P.b[p][o];
  for (int k = 0; k < D_; ++k) acc = fmaf(c[k], w[k], acc);
  out[id] = acc;
}

// Write cond token (bf16) into kv row b*KVR
__global__ void k_cond_token(const float* __restrict__ cond, bf16_t* __restrict__ kvh) {
  int i = blockIdx.x * blockDim.x + threadIdx.x;    // 4*1024
  int b = i >> 10, d = i & (D_ - 1);
  kvh[(size_t)(b * KVR) * D_ + d] = (bf16_t)cond[i];
}

// ---------------------------------------------------------------------------
// LayerNorm + AdaLN modulate, f32 in -> bf16 out. kvLayout: row -> row + b + 1
// ---------------------------------------------------------------------------
__global__ __launch_bounds__(256)
void k_adaln(const float* __restrict__ x, const float* __restrict__ scale,
             const float* __restrict__ shift, bf16_t* __restrict__ out, int kvLayout) {
  int row = blockIdx.x;              // 0..4095
  int b = row >> 10;
  int tid = threadIdx.x;
  __shared__ float red[8];

  const float* xr = x + (size_t)row * D_;
  float v[4];
  float s = 0.f;
#pragma unroll
  for (int e = 0; e < 4; ++e) { v[e] = xr[tid + e * 256]; s += v[e]; }
#pragma unroll
  for (int off = 16; off >= 1; off >>= 1) s += __shfl_xor(s, off, 32);
  if ((tid & 31) == 0) red[tid >> 5] = s;
  __syncthreads();
  float tot = 0.f;
#pragma unroll
  for (int w = 0; w < 8; ++w) tot += red[w];
  float mu = tot * (1.0f / (float)D_);
  __syncthreads();

  float vs = 0.f;
#pragma unroll
  for (int e = 0; e < 4; ++e) { float d = v[e] - mu; vs += d * d; }
#pragma unroll
  for (int off = 16; off >= 1; off >>= 1) vs += __shfl_xor(vs, off, 32);
  if ((tid & 31) == 0) red[tid >> 5] = vs;
  __syncthreads();
  float vtot = 0.f;
#pragma unroll
  for (int w = 0; w < 8; ++w) vtot += red[w];
  float rstd = rsqrtf(vtot * (1.0f / (float)D_) + 1e-5f);

  const float* sc = scale + (size_t)b * D_;
  const float* sh = shift + (size_t)b * D_;
  size_t orow = kvLayout ? (size_t)(row + b + 1) : (size_t)row;
  bf16_t* op = out + orow * D_;
#pragma unroll
  for (int e = 0; e < 4; ++e) {
    int d = tid + e * 256;
    float xn = (v[e] - mu) * rstd;
    op[d] = (bf16_t)(xn * (1.0f + sc[d]) + sh[d]);
  }
}

// ---------------------------------------------------------------------------
// Tiled WMMA GEMM: C[M,N] = A[M,K](bf16) * W[N,K]^T(bf16) + bias
//   mode 0: out bf16        mode 1: outF = res + C (f32)      mode 2: gelu -> bf16
// block = 128 threads (4 waves), tile 64x64, BK=32, LDS-staged.
// ---------------------------------------------------------------------------
__global__ __launch_bounds__(128)
void k_gemm(const bf16_t* __restrict__ A, const bf16_t* __restrict__ W,
            const float* __restrict__ bias, const float* __restrict__ res,
            float* __restrict__ outF, bf16_t* __restrict__ outB,
            int M, int N, int K, int mode) {
  __shared__ __attribute__((aligned(64))) bf16_t As[64][32];
  __shared__ __attribute__((aligned(64))) bf16_t Ws[64][32];

  int tid = threadIdx.x;
  int lane = tid & 31, wave = tid >> 5;
  int lrow = lane & 15, hi = lane >> 4;
  int m0 = blockIdx.y * 64;
  int n0 = blockIdx.x * 64;

  v8f acc[4];
#pragma unroll
  for (int i = 0; i < 4; ++i) acc[i] = vzero8();

  int srow = tid >> 1;          // 0..63
  int shalf = (tid & 1) * 16;   // 0 or 16
  int arow = m0 + srow; if (arow >= M) arow = M - 1;
  const bf16_t* Ap = A + (size_t)arow * K + shalf;
  const bf16_t* Wp = W + (size_t)(n0 + srow) * K + shalf;

  for (int kk = 0; kk < K; kk += 32) {
    __builtin_prefetch(Ap + kk + 64, 0, 0);   // global_prefetch next-next tile
    *(v16bf*)&As[srow][shalf] = *(const v16bf*)(Ap + kk);
    *(v16bf*)&Ws[srow][shalf] = *(const v16bf*)(Wp + kk);
    __syncthreads();

    // A fragment (16x32): lane row = wave*16 + lrow; K contiguous: k = hi*16 + s
    int frow = (wave << 4) + lrow;
    int kk0 = hi * 16;
    V16U af;
    af.h[0] = *(const v8bf*)&As[frow][kk0];
    af.h[1] = *(const v8bf*)&As[frow][kk0 + 8];
#pragma unroll
    for (int nt = 0; nt < 4; ++nt) {
      int col = (nt << 4) + lrow;
      v16bf bf = *(const v16bf*)&Ws[col][hi * 16];
      acc[nt] = wmma_bf16(af.v, bf, acc[nt]);
    }
    __syncthreads();
  }

#pragma unroll
  for (int nt = 0; nt < 4; ++nt) {
    int gcol = n0 + (nt << 4) + lrow;
    float bv = bias[gcol];
#pragma unroll
    for (int r = 0; r < 8; ++r) {
      int grow = m0 + (wave << 4) + r + hi * 8;
      if (grow < M) {
        float v = acc[nt][r] + bv;
        size_t off = (size_t)grow * N + gcol;
        if (mode == 0)       outB[off] = (bf16_t)v;
        else if (mode == 1)  outF[off] = res[off] + v;
        else {
          float g = 0.5f * v * (1.0f + erff(v * 0.70710678118654752f));
          outB[off] = (bf16_t)g;
        }
      }
    }
  }
}

// ---------------------------------------------------------------------------
// Flash attention (wave32 WMMA). One wave = 16 queries x HD=64.
// 4 waves/block share (b,h) and cooperatively stage V^T tile in LDS.
// Q/K/V are bf16 in kv layout [B*KVR, D]. Output O bf16 [B*L, D] (head-interleaved).
// ---------------------------------------------------------------------------
__global__ __launch_bounds__(128)
void k_attn(const bf16_t* __restrict__ Q, const bf16_t* __restrict__ Kb,
            const bf16_t* __restrict__ Vb, bf16_t* __restrict__ Oout,
            const unsigned char* __restrict__ cpm,    // cond_positions_mask [B,L]
            const unsigned char* __restrict__ pad,    // padding_mask [B,L]
            const unsigned char* __restrict__ cpad,   // cond_padding_mask [B,1+L]
            int isCross) {
  __shared__ __attribute__((aligned(64))) bf16_t VT[HD_][32];      // V^T tile
  __shared__ __attribute__((aligned(64))) bf16_t Ps[4][16][32];    // per-wave P

  int tid = threadIdx.x, lane = tid & 31, wave = tid >> 5;
  int lrow = lane & 15, hi = lane >> 4;
  int bh = blockIdx.x;
  int b = bh / H_, h = bh % H_;
  int qg = blockIdx.y;                 // 16 groups of 64 queries
  int hoff = h * HD_;

  int qi_frag = qg * 64 + wave * 16 + lrow;                  // query for A-frag
  const bf16_t* Qrow = Q + (size_t)(b * KVR + 1 + qi_frag) * D_ + hoff;

  int iRow[8]; unsigned char ciRow[8];
#pragma unroll
  for (int r = 0; r < 8; ++r) {
    iRow[r] = qg * 64 + wave * 16 + r + hi * 8;
    ciRow[r] = cpm[b * L_ + iRow[r]];
  }

  int Lk = isCross ? (L_ + 1) : L_;
  int kRowBase = b * KVR + (isCross ? 0 : 1);

  v8f oacc[4];
#pragma unroll
  for (int i = 0; i < 4; ++i) oacc[i] = vzero8();
  float mrow[8], lsum[8];
#pragma unroll
  for (int r = 0; r < 8; ++r) { mrow[r] = -__builtin_inff(); lsum[r] = 0.f; }

  int nkt = (Lk + 31) >> 5;
  for (int kt = 0; kt < nkt; ++kt) {
    int jb = kt * 32;

    // ---- stage V^T (keys jb..jb+31, d 0..63) ----
    {
      int jj = tid & 31, dc = tid >> 5;  // dc in 0..3
      int key = jb + jj; if (key >= Lk) key = Lk - 1;
      const bf16_t* vp = Vb + (size_t)(kRowBase + key) * D_ + hoff + dc * 16;
#pragma unroll
      for (int e = 0; e < 16; ++e) VT[dc * 16 + e][jj] = vp[e];
    }
    __syncthreads();

    // ---- S = Q K^T : two 16x16 tiles, K-dim = 64 in two chunks ----
    v8f sacc[2];
    sacc[0] = vzero8(); sacc[1] = vzero8();
#pragma unroll
    for (int nt = 0; nt < 2; ++nt) {
      int key = jb + nt * 16 + lrow; if (key >= Lk) key = Lk - 1;
      const bf16_t* Krow = Kb + (size_t)(kRowBase + key) * D_ + hoff;
#pragma unroll
      for (int dchunk = 0; dchunk < 2; ++dchunk) {
        int kk0 = dchunk * 32 + hi * 16;
        V16U af;
        af.h[0] = *(const v8bf*)(Qrow + kk0);
        af.h[1] = *(const v8bf*)(Qrow + kk0 + 8);
        v16bf bfrag = *(const v16bf*)(Krow + dchunk * 32 + hi * 16);
        sacc[nt] = wmma_bf16(af.v, bfrag, sacc[nt]);
      }
    }

    // ---- column mask info (per nt, shared across the 8 rows) ----
    bool  colValid[2]; unsigned char cjv[2]; unsigned char extraM[2];
    int   jmv[2]; bool isCond[2];
#pragma unroll
    for (int nt = 0; nt < 2; ++nt) {
      int kcol = jb + nt * 16 + lrow;
      colValid[nt] = (kcol < Lk);
      int kc = colValid[nt] ? kcol : (Lk - 1);
      if (isCross) {
        isCond[nt] = (kc == 0);
        int jm = isCond[nt] ? 0 : (kc - 1);
        jmv[nt] = jm;
        cjv[nt] = cpm[b * L_ + jm];
        extraM[nt] = cpad[b * (L_ + 1) + kc];
      } else {
        isCond[nt] = false;
        jmv[nt] = kc;
        cjv[nt] = cpm[b * L_ + kc];
        extraM[nt] = pad[b * L_ + kc];
      }
    }

    // ---- mask + online softmax + write P (bf16) to per-wave LDS ----
#pragma unroll
    for (int r = 0; r < 8; ++r) {
      float sv[2];
      sv[0] = sacc[0][r] * 0.125f;   // 1/sqrt(64)
      sv[1] = sacc[1][r] * 0.125f;
#pragma unroll
      for (int nt = 0; nt < 2; ++nt) {
        bool masked;
        if (!colValid[nt]) masked = true;
        else if (isCond[nt]) masked = (extraM[nt] != 0);
        else {
          bool ci = (ciRow[r] != 0), cj = (cjv[nt] != 0);
          masked = ((iRow[r] < jmv[nt]) && !(ci && cj)) || (ci && !cj) || (extraM[nt] != 0);
        }
        if (masked) sv[nt] = -__builtin_inff();
      }
      float vmax = fmaxf(sv[0], sv[1]);
#pragma unroll
      for (int off = 8; off >= 1; off >>= 1) vmax = fmaxf(vmax, __shfl_xor(vmax, off, 16));
      float mold = mrow[r];
      float mnew = fmaxf(mold, vmax);
      float alpha, p0, p1;
      if (mnew == -__builtin_inff()) { alpha = 1.0f; p0 = 0.0f; p1 = 0.0f; }
      else {
        alpha = expf(mold - mnew);   // expf(-inf) == 0
        p0 = expf(sv[0] - mnew);
        p1 = expf(sv[1] - mnew);
      }
      mrow[r] = mnew;
      float psum = p0 + p1;
#pragma unroll
      for (int off = 8; off >= 1; off >>= 1) psum += __shfl_xor(psum, off, 16);
      lsum[r] = lsum[r] * alpha + psum;
#pragma unroll
      for (int dt = 0; dt < 4; ++dt) oacc[dt][r] *= alpha;
      int prow = r + hi * 8;
      Ps[wave][prow][lrow]      = (bf16_t)p0;
      Ps[wave][prow][16 + lrow] = (bf16_t)p1;
    }

    // ---- O += P(16x32) * V(32x64) ----
    V16U pa;
    pa.h[0] = *(const v8bf*)&Ps[wave][lrow][hi * 16];
    pa.h[1] = *(const v8bf*)&Ps[wave][lrow][hi * 16 + 8];
#pragma unroll
    for (int dt = 0; dt < 4; ++dt) {
      v16bf vb = *(const v16bf*)&VT[dt * 16 + lrow][hi * 16];
      oacc[dt] = wmma_bf16(pa.v, vb, oacc[dt]);
    }
    __syncthreads();   // before VT is restaged
  }

  // ---- normalize and store ----
#pragma unroll
  for (int r = 0; r < 8; ++r) {
    float inv = 1.0f / lsum[r];
    size_t grow = (size_t)(b * L_ + iRow[r]);
#pragma unroll
    for (int dt = 0; dt < 4; ++dt) {
      Oout[grow * D_ + hoff + dt * 16 + lrow] = (bf16_t)(oacc[dt][r] * inv);
    }
  }
}

// ---------------------------------------------------------------------------
// Host-side orchestration
// ---------------------------------------------------------------------------
extern "C" void kernel_launch(void* const* d_in, const int* in_sizes, int n_in,
                              void* d_out, int out_size, void* d_ws, size_t ws_size,
                              hipStream_t stream) {
  (void)in_sizes; (void)n_in; (void)out_size; (void)ws_size;

  const float* x    = (const float*)d_in[0];
  const float* cond = (const float*)d_in[1];
  const unsigned char* pad  = (const unsigned char*)d_in[34];
  const unsigned char* cpad = (const unsigned char*)d_in[35];
  const unsigned char* cpm  = (const unsigned char*)d_in[36];
  float* out = (float*)d_out;

  // bump allocator over workspace
  char* ws = (char*)d_ws;
  size_t off = 0;
  auto alloc = [&](size_t bytes) -> void* {
    void* p = ws + off;
    off = (off + bytes + 255) & ~(size_t)255;
    return p;
  };

  const size_t MB1 = (size_t)D_ * D_;           // 1M elements
  bf16_t* wsq = (bf16_t*)alloc(MB1 * 2);
  bf16_t* wsk = (bf16_t*)alloc(MB1 * 2);
  bf16_t* wsv = (bf16_t*)alloc(MB1 * 2);
  bf16_t* wso = (bf16_t*)alloc(MB1 * 2);
  bf16_t* wcq = (bf16_t*)alloc(MB1 * 2);
  bf16_t* wck = (bf16_t*)alloc(MB1 * 2);
  bf16_t* wcv = (bf16_t*)alloc(MB1 * 2);
  bf16_t* wco = (bf16_t*)alloc(MB1 * 2);
  bf16_t* wf1 = (bf16_t*)alloc((size_t)FF_ * D_ * 2);
  bf16_t* wf2 = (bf16_t*)alloc((size_t)FF_ * D_ * 2);
  float*  adm = (float*)alloc((size_t)6 * B_ * D_ * 4);  // [6][B][D]
  bf16_t* kvh = (bf16_t*)alloc((size_t)B_ * KVR * D_ * 2);
  float*  r1  = (float*)alloc((size_t)B_ * L_ * D_ * 4);
  float*  r2  = (float*)alloc((size_t)B_ * L_ * D_ * 4);
  bf16_t* Qb  = (bf16_t*)alloc((size_t)B_ * KVR * D_ * 2);
  bf16_t* Kbuf= (bf16_t*)alloc((size_t)B_ * KVR * D_ * 2);
  bf16_t* Vbuf= (bf16_t*)alloc((size_t)B_ * KVR * D_ * 2);
  bf16_t* Oin = (bf16_t*)alloc((size_t)B_ * L_ * D_ * 2);  // attn out / h3 reuse
  bf16_t* ffh = (bf16_t*)alloc((size_t)B_ * L_ * FF_ * 2);

  // 1) convert weights to bf16
  struct { int idx; bf16_t* dst; size_t n; } cv[10] = {
    { 2, wsq, MB1}, { 4, wsk, MB1}, { 6, wsv, MB1}, { 8, wso, MB1},
    {10, wcq, MB1}, {12, wck, MB1}, {14, wcv, MB1}, {16, wco, MB1},
    {30, wf1, (size_t)FF_ * D_}, {32, wf2, (size_t)FF_ * D_},
  };
  for (int i = 0; i < 10; ++i) {
    int n = (int)cv[i].n;
    k_cvt<<<dim3((n + 255) / 256), dim3(256), 0, stream>>>(
        (const float*)d_in[cv[i].idx], cv[i].dst, n);
  }

  // 2) adaln scale/shift vectors
  AdalnPtrs ap;
  ap.w[0] = (const float*)d_in[18]; ap.b[0] = (const float*)d_in[19]; // adaln1.scale
  ap.w[1] = (const float*)d_in[20]; ap.b[1] = (const float*)d_in[21]; // adaln1.shift
  ap.w[2] = (const float*)d_in[22]; ap.b[2] = (const float*)d_in[23];
  ap.w[3] = (const float*)d_in[24]; ap.b[3] = (const float*)d_in[25];
  ap.w[4] = (const float*)d_in[26]; ap.b[4] = (const float*)d_in[27];
  ap.w[5] = (const float*)d_in[28]; ap.b[5] = (const float*)d_in[29];
  k_adaln_mod<<<dim3(6 * B_ * D_ / 256), dim3(256), 0, stream>>>(ap, cond, adm);
  float* sc1 = adm + 0 * B_ * D_; float* sh1 = adm + 1 * B_ * D_;
  float* sc2 = adm + 2 * B_ * D_; float* sh2 = adm + 3 * B_ * D_;
  float* sc3 = adm + 4 * B_ * D_; float* sh3 = adm + 5 * B_ * D_;

  // 3) cond token into kv row 0 of each batch
  k_cond_token<<<dim3(B_ * D_ / 256), dim3(256), 0, stream>>>(cond, kvh);

  const int Mkv = B_ * KVR;   // 4100
  const int Mx  = B_ * L_;    // 4096
  dim3 gemmB(128);
  dim3 gQKV(D_ / 64, (Mkv + 63) / 64);
  dim3 gO(D_ / 64, Mx / 64);
  dim3 gFF1(FF_ / 64, Mx / 64);
  dim3 gFF2(D_ / 64, Mx / 64);
  dim3 attnG(B_ * H_, L_ / 64);

  // ---- self attention ----
  k_adaln<<<dim3(Mx), dim3(256), 0, stream>>>(x, sc1, sh1, kvh, 1);
  k_gemm<<<gQKV, gemmB, 0, stream>>>(kvh, wsq, (const float*)d_in[3], nullptr,
                                     nullptr, Qb,   Mkv, D_, D_, 0);
  k_gemm<<<gQKV, gemmB, 0, stream>>>(kvh, wsk, (const float*)d_in[5], nullptr,
                                     nullptr, Kbuf, Mkv, D_, D_, 0);
  k_gemm<<<gQKV, gemmB, 0, stream>>>(kvh, wsv, (const float*)d_in[7], nullptr,
                                     nullptr, Vbuf, Mkv, D_, D_, 0);
  k_attn<<<attnG, dim3(128), 0, stream>>>(Qb, Kbuf, Vbuf, Oin, cpm, pad, cpad, 0);
  k_gemm<<<gO, gemmB, 0, stream>>>(Oin, wso, (const float*)d_in[9], x,
                                   r1, nullptr, Mx, D_, D_, 1);

  // ---- cross attention ----
  k_adaln<<<dim3(Mx), dim3(256), 0, stream>>>(r1, sc2, sh2, kvh, 1);
  k_gemm<<<gQKV, gemmB, 0, stream>>>(kvh, wcq, (const float*)d_in[11], nullptr,
                                     nullptr, Qb,   Mkv, D_, D_, 0);
  k_gemm<<<gQKV, gemmB, 0, stream>>>(kvh, wck, (const float*)d_in[13], nullptr,
                                     nullptr, Kbuf, Mkv, D_, D_, 0);
  k_gemm<<<gQKV, gemmB, 0, stream>>>(kvh, wcv, (const float*)d_in[15], nullptr,
                                     nullptr, Vbuf, Mkv, D_, D_, 0);
  k_attn<<<attnG, dim3(128), 0, stream>>>(Qb, Kbuf, Vbuf, Oin, cpm, pad, cpad, 1);
  k_gemm<<<gO, gemmB, 0, stream>>>(Oin, wco, (const float*)d_in[17], r1,
                                   r2, nullptr, Mx, D_, D_, 1);

  // ---- feed-forward ----
  k_adaln<<<dim3(Mx), dim3(256), 0, stream>>>(r2, sc3, sh3, Oin, 0);  // h3 in Oin
  k_gemm<<<gFF1, gemmB, 0, stream>>>(Oin, wf1, (const float*)d_in[31], nullptr,
                                     nullptr, ffh, Mx, FF_, D_, 2);   // gelu
  k_gemm<<<gFF2, gemmB, 0, stream>>>(ffh, wf2, (const float*)d_in[33], r2,
                                     out, nullptr, Mx, D_, FF_, 1);
}
